// PCTransformer_37151467111130
// MI455X (gfx1250) — matrix operand order repaired and checked by reading, not compiled, not run on hardware
//
#include <hip/hip_runtime.h>
#include <stdint.h>

// ---------------------------------------------------------------------------
// Problem constants (from reference)
// ---------------------------------------------------------------------------
enum { B_ = 16, N_ = 512, C_ = 768, H_ = 6, HD_ = 128, DEPTH_ = 6,
       K_ = 8, HID_ = 1536, OUT_ = 1024, M_ = B_ * N_ };

typedef unsigned short u16;
typedef __attribute__((ext_vector_type(16))) __bf16        v16bf;
typedef __attribute__((ext_vector_type(8)))  float         v8f;
typedef __attribute__((ext_vector_type(4)))  unsigned int  u32x4;
typedef __attribute__((ext_vector_type(4)))  int           v4i;

__device__ __forceinline__ u16 f2bf(float f) {
  unsigned int u = __float_as_uint(f);          // RNE fp32 -> bf16
  u += 0x7FFFu + ((u >> 16) & 1u);
  return (u16)(u >> 16);
}

// ---------------------------------------------------------------------------
// gfx1250 async global->LDS copy (ASYNCcnt) with sync fallback
// ---------------------------------------------------------------------------
#if defined(__has_builtin)
#  if __has_builtin(__builtin_amdgcn_global_load_async_to_lds_b128) && \
      __has_builtin(__builtin_amdgcn_s_wait_asynccnt)
#    define PCT_ASYNC 1
#  endif
#endif
#ifndef PCT_ASYNC
#  define PCT_ASYNC 0
#endif

__device__ __forceinline__ void copy16(const u16* g, u16* l) {
#if PCT_ASYNC
  __builtin_amdgcn_global_load_async_to_lds_b128(
      (__attribute__((address_space(1))) v4i*)(uintptr_t)(const void*)g,
      (__attribute__((address_space(3))) v4i*)(uintptr_t)(void*)l, 0, 0);
#else
  *(u32x4*)l = *(const u32x4*)g;
#endif
}
__device__ __forceinline__ void wait_copies() {
#if PCT_ASYNC
  __builtin_amdgcn_s_wait_asynccnt(0);
#endif
}

// ---------------------------------------------------------------------------
// Core batched GEMM:  C[m,n] = act( scale * sum_k A[m,k]*B[n,k] + bias[n] )
//   A, B: bf16, K-contiguous (lda/ldb in elements).  C: fp32 and/or bf16.
//   Block tile 128x128, K-step 32, 8 waves (4x2), each wave 32x64:
//   8 x v_wmma_f32_16x16x32_bf16 per wave per K-step, double-buffered LDS
//   fed by GLOBAL_LOAD_ASYNC_TO_LDS_B128.
// ---------------------------------------------------------------------------
#define BM 128
#define BN 128
#define BK 32

__global__ __launch_bounds__(256)
void pct_gemm_wmma(const u16* __restrict__ A, int lda, long long aStrB, long long aStrH,
                   const u16* __restrict__ Bq, int ldb, long long bStrB, long long bStrH,
                   float* __restrict__ Cf, u16* __restrict__ Cbf,
                   int ldc, long long cStrB, long long cStrH,
                   const float* __restrict__ bias,
                   int Kdim, int nH, float scale, int act, int accum)
{
  __shared__ __align__(16) u16 As[2][BM][BK + 8];
  __shared__ __align__(16) u16 Bs[2][BN][BK + 8];

  const int z  = blockIdx.z;
  const int bb = z / nH, hh = z % nH;
  A  += (long long)bb * aStrB + (long long)hh * aStrH;
  Bq += (long long)bb * bStrB + (long long)hh * bStrH;
  if (Cf)  Cf  += (long long)bb * cStrB + (long long)hh * cStrH;
  if (Cbf) Cbf += (long long)bb * cStrB + (long long)hh * cStrH;

  const int rowBase = blockIdx.y * BM;
  const int colBase = blockIdx.x * BN;
  const int tid  = threadIdx.x;
  const int lane = tid & 31;
  const int wave = tid >> 5;
  const int wm   = wave >> 1;     // 0..3 : 32-row band
  const int wn   = wave & 1;      // 0..1 : 64-col band
  const int half = lane >> 4;
  const int l16  = lane & 15;

  // stage one 128x32 bf16 tile of A and B into LDS buffer `buf`
  auto stage = [&](int buf, int kb) {
    #pragma unroll
    for (int i = 0; i < 2; ++i) {
      int c  = tid + i * 256;      // 512 16B-chunks per tile
      int m  = c >> 2;
      int ke = (c & 3) * 8;
      copy16(&A [(size_t)(rowBase + m) * lda + kb + ke], &As[buf][m][ke]);
      copy16(&Bq[(size_t)(colBase + m) * ldb + kb + ke], &Bs[buf][m][ke]);
    }
  };

  v8f acc[2][4];
  #pragma unroll
  for (int i = 0; i < 2; ++i)
    #pragma unroll
    for (int j = 0; j < 4; ++j)
      #pragma unroll
      for (int r = 0; r < 8; ++r) acc[i][j][r] = 0.0f;

  const int nk = Kdim / BK;
  stage(0, 0);
  for (int t = 0; t < nk; ++t) {
    const int cur = t & 1;
    wait_copies();          // our async writes into buf `cur` are done
    __syncthreads();        // everyone's writes done; prev reads of buf cur^1 done
    if (t + 1 < nk) stage(cur ^ 1, (t + 1) * BK);   // overlap with WMMA below

    union Frag { u32x4 q[2]; v16bf v; };
    Frag fa[2], fb[4];
    #pragma unroll
    for (int s = 0; s < 2; ++s) {               // A: M = lane%16
      int r = wm * 32 + s * 16 + l16;
      fa[s].q[0] = *(const u32x4*)&As[cur][r][8 * half];        // K 8h..8h+7
      fa[s].q[1] = *(const u32x4*)&As[cur][r][16 + 8 * half];   // K 16+8h..+7
    }
    #pragma unroll
    for (int j = 0; j < 4; ++j) {               // B: N = lane%16
      int n = wn * 64 + j * 16 + l16;
      fb[j].q[0] = *(const u32x4*)&Bs[cur][n][8 * half];        // K 8h..8h+7
      fb[j].q[1] = *(const u32x4*)&Bs[cur][n][16 + 8 * half];   // K 16+8h..+7
    }
    #pragma unroll
    for (int i = 0; i < 2; ++i)
      #pragma unroll
      for (int j = 0; j < 4; ++j)
        acc[i][j] = __builtin_amdgcn_wmma_f32_16x16x32_bf16(
            false, fa[i].v, false, fb[j].v, (short)0, acc[i][j], false, false);
  }

  // epilogue: D layout VGPR r -> M = r + 8*half, N = lane%16
  #pragma unroll
  for (int i = 0; i < 2; ++i) {
    #pragma unroll
    for (int j = 0; j < 4; ++j) {
      int col = colBase + wn * 64 + j * 16 + l16;
      float bv = bias ? bias[col] : 0.0f;
      #pragma unroll
      for (int r = 0; r < 8; ++r) {
        int row = rowBase + wm * 32 + i * 16 + half * 8 + r;
        float v = acc[i][j][r] * scale + bv;
        if (act == 1)      v = (v >= 0.0f) ? v : 0.2f * v;                         // lrelu
        else if (act == 2) v = 0.5f * v * (1.0f + erff(v * 0.70710678118654752f)); // gelu
        size_t off = (size_t)row * ldc + col;
        if (accum) v += Cf[off];
        if (Cf)  Cf[off]  = v;
        if (Cbf) Cbf[off] = f2bf(v);
      }
    }
  }
}

// ---------------------------------------------------------------------------
// Support kernels
// ---------------------------------------------------------------------------
__global__ void pct_cast(const float* __restrict__ in, u16* __restrict__ out, size_t n)
{
  size_t t = (size_t)blockIdx.x * blockDim.x + threadIdx.x;
  if (t < n) out[t] = f2bf(in[t]);
}

// pe_w1 (128,3) fp32 -> (128,32) bf16, zero K-padded
__global__ void pct_pad_w1(const float* __restrict__ w, u16* __restrict__ out)
{
  int t = blockIdx.x * blockDim.x + threadIdx.x;
  if (t >= 128 * 32) return;
  int o = t >> 5, k = t & 31;
  out[t] = (k < 3) ? f2bf(w[o * 3 + k]) : (u16)0;
}

// (B, Cin, N) fp32 -> (B*N, Kpad) bf16, zero K-padded token-major
__global__ void pct_transpose_pad(const float* __restrict__ in, u16* __restrict__ out,
                                  int Cin, int Kpad)
{
  size_t t = (size_t)blockIdx.x * blockDim.x + threadIdx.x;
  size_t total = (size_t)M_ * Kpad;
  if (t >= total) return;
  int k = (int)(t % Kpad);
  size_t m = t / Kpad;
  int b = (int)(m / N_), n = (int)(m % N_);
  out[t] = (k < Cin) ? f2bf(in[((size_t)b * Cin + k) * N_ + n]) : (u16)0;
}

// v = lrelu(x/sqrt(1+eps)*g[o]+b[o]) -> bf16 (consumed only by GEMM)
__global__ void pct_bn_lrelu(const float* __restrict__ x, u16* __restrict__ outb,
                             const float* __restrict__ g, const float* __restrict__ b, int L)
{
  size_t t = (size_t)blockIdx.x * blockDim.x + threadIdx.x;
  size_t total = (size_t)M_ * L;
  if (t >= total) return;
  int o = (int)(t % L);
  float inv = rsqrtf(1.0f + 1e-5f);
  float v = x[t] * inv * g[o] + b[o];
  v = (v >= 0.0f) ? v : 0.2f * v;
  outb[t] = f2bf(v);
}

__global__ void pct_add(float* __restrict__ out, const float* __restrict__ a,
                        const float* __restrict__ b, size_t n, u16* __restrict__ outb)
{
  size_t t = (size_t)blockIdx.x * blockDim.x + threadIdx.x;
  if (t >= n) return;
  float v = a[t] + b[t];
  out[t] = v;
  if (outb) outb[t] = f2bf(v);
}

__global__ __launch_bounds__(256)
void pct_ln_row(const float* __restrict__ in, float* __restrict__ outf,
                u16* __restrict__ outb,
                const float* __restrict__ g, const float* __restrict__ b, int L)
{
  __shared__ float s1[256], s2[256];
  size_t base = (size_t)blockIdx.x * L;
  float sum = 0.0f, sq = 0.0f;
  for (int c = threadIdx.x; c < L; c += 256) {
    float v = in[base + c]; sum += v; sq += v * v;
  }
  s1[threadIdx.x] = sum; s2[threadIdx.x] = sq;
  __syncthreads();
  for (int o = 128; o > 0; o >>= 1) {
    if (threadIdx.x < o) { s1[threadIdx.x] += s1[threadIdx.x + o];
                           s2[threadIdx.x] += s2[threadIdx.x + o]; }
    __syncthreads();
  }
  float mean = s1[0] / L;
  float var  = s2[0] / L - mean * mean;
  float rstd = rsqrtf(var + 1e-5f);
  for (int c = threadIdx.x; c < L; c += 256) {
    float v = (in[base + c] - mean) * rstd * g[c] + b[c];
    if (outf) outf[base + c] = v;
    if (outb) outb[base + c] = f2bf(v);
  }
}

__global__ __launch_bounds__(256)
void pct_softmax_row(float* __restrict__ x, u16* __restrict__ outb, int L)
{
  __shared__ float s[256];
  size_t base = (size_t)blockIdx.x * L;
  float m = -3.4e38f;
  for (int c = threadIdx.x; c < L; c += 256) m = fmaxf(m, x[base + c]);
  s[threadIdx.x] = m; __syncthreads();
  for (int o = 128; o > 0; o >>= 1) {
    if (threadIdx.x < o) s[threadIdx.x] = fmaxf(s[threadIdx.x], s[threadIdx.x + o]);
    __syncthreads();
  }
  m = s[0]; __syncthreads();
  float sum = 0.0f;
  for (int c = threadIdx.x; c < L; c += 256) {
    float e = __expf(x[base + c] - m); x[base + c] = e; sum += e;
  }
  s[threadIdx.x] = sum; __syncthreads();
  for (int o = 128; o > 0; o >>= 1) {
    if (threadIdx.x < o) s[threadIdx.x] += s[threadIdx.x + o];
    __syncthreads();
  }
  float inv = 1.0f / s[0];
  for (int c = threadIdx.x; c < L; c += 256)
    outb[base + c] = f2bf(x[base + c] * inv);
}

__global__ void pct_knn_topk(const float* __restrict__ coor, int* __restrict__ idx)
{
  int t = blockIdx.x * blockDim.x + threadIdx.x;
  if (t >= M_) return;
  int b = t / N_, n = t % N_;
  const float* cb = coor + (size_t)b * 3 * N_;
  float px = cb[n], py = cb[N_ + n], pz = cb[2 * N_ + n];
  float sqn = px * px + py * py + pz * pz;
  float bd[K_]; int bi[K_];
  #pragma unroll
  for (int k = 0; k < K_; ++k) { bd[k] = 3.4e38f; bi[k] = -1; }
  for (int m = 0; m < N_; ++m) {
    float qx = cb[m], qy = cb[N_ + m], qz = cb[2 * N_ + m];
    float d2 = sqn + (qx * qx + qy * qy + qz * qz)
             - 2.0f * (px * qx + py * qy + pz * qz);
    float d = sqrtf(fmaxf(d2, 0.0f));
    if (d < bd[K_ - 1]) {      // strict <  => stable ties, earliest index wins
      int p = K_ - 1;
      while (p > 0 && d < bd[p - 1]) { bd[p] = bd[p - 1]; bi[p] = bi[p - 1]; --p; }
      bd[p] = d; bi[p] = m;
    }
  }
  #pragma unroll
  for (int k = 0; k < K_; ++k)
    idx[(size_t)b * K_ * N_ + k * N_ + n] = bi[k];
}

// knn_f[b,n,o] = max_k lrelu( t1[b, idx[b,k,n], o] + t2[b,n,o] ) -> bf16
__global__ void pct_knn_gather_max(const float* __restrict__ t1, const float* __restrict__ t2,
                                   const int* __restrict__ idx, u16* __restrict__ outb)
{
  size_t t = (size_t)blockIdx.x * blockDim.x + threadIdx.x;
  size_t total = (size_t)M_ * C_;
  if (t >= total) return;
  int o = (int)(t % C_);
  size_t m = t / C_;
  int b = (int)(m / N_), n = (int)(m % N_);
  float c = t2[t];
  float best = -3.4e38f;
  #pragma unroll
  for (int k = 0; k < K_; ++k) {
    int g = idx[(size_t)b * K_ * N_ + k * N_ + n];
    float v = t1[((size_t)b * N_ + g) * C_ + o] + c;
    v = (v >= 0.0f) ? v : 0.2f * v;
    best = fmaxf(best, v);
  }
  outb[t] = f2bf(best);
}

// vT[(b*H+h), d, m] = V[b, m, h, d]  (from qkv bf16 mirror)
__global__ void pct_vt(const u16* __restrict__ qkvbf, u16* __restrict__ vT)
{
  size_t t = (size_t)blockIdx.x * blockDim.x + threadIdx.x;
  size_t total = (size_t)B_ * H_ * HD_ * N_;
  if (t >= total) return;
  int m = (int)(t % N_);
  size_t r = t / N_;
  int d = (int)(r % HD_);
  size_t bh = r / HD_;
  int b = (int)(bh / H_), h = (int)(bh % H_);
  vT[t] = qkvbf[(size_t)(b * N_ + m) * 3 * C_ + 2 * C_ + h * HD_ + d];
}

__global__ void pct_max_over_n(const float* __restrict__ g2, float* __restrict__ gf)
{
  int t = blockIdx.x * blockDim.x + threadIdx.x;
  if (t >= B_ * OUT_) return;
  int b = t / OUT_, o = t % OUT_;
  float best = -3.4e38f;
  for (int n = 0; n < N_; ++n)
    best = fmaxf(best, g2[((size_t)b * N_ + n) * OUT_ + o]);
  gf[t] = best;
}

// wd[o,c] = knn_w[o, C+c] - knn_w[o, c]  -> bf16 (exact edge-MLP split)
__global__ void pct_wd_prep(const float* __restrict__ knn_w, u16* __restrict__ wd)
{
  int t = blockIdx.x * blockDim.x + threadIdx.x;
  if (t >= C_ * C_) return;
  int o = t / C_, c = t % C_;
  wd[t] = f2bf(knn_w[(size_t)o * 2 * C_ + C_ + c] - knn_w[(size_t)o * 2 * C_ + c]);
}

// ---------------------------------------------------------------------------
// Host-side helpers
// ---------------------------------------------------------------------------
static inline dim3 blk1(size_t n) { return dim3((unsigned)((n + 255) / 256)); }

static void run_gemm(hipStream_t s, const u16* A, int lda, const u16* Bp, int ldb,
                     float* Cf, u16* Cbf, int ldc, const float* bias,
                     int M, int Nd, int Kd, int act = 0, int accum = 0, float scale = 1.0f,
                     int batches = 1, int nH = 1,
                     long long aB = 0, long long aH = 0,
                     long long bB = 0, long long bH = 0,
                     long long cB = 0, long long cH = 0)
{
  dim3 g(Nd / BN, M / BM, batches);
  pct_gemm_wmma<<<g, 256, 0, s>>>(A, lda, aB, aH, Bp, ldb, bB, bH,
                                  Cf, Cbf, ldc, cB, cH, bias, Kd, nH, scale, act, accum);
}

static void castw(hipStream_t s, const float* src, u16* dst, size_t n)
{
  pct_cast<<<blk1(n), 256, 0, s>>>(src, dst, n);
}

extern "C" void kernel_launch(void* const* d_in, const int* in_sizes, int n_in,
                              void* d_out, int out_size, void* d_ws, size_t ws_size,
                              hipStream_t stream)
{
  const float* coor   = (const float*)d_in[0];
  const float* f_in   = (const float*)d_in[1];
  const float* pe_w1  = (const float*)d_in[2];
  const float* pe_b1  = (const float*)d_in[3];
  const float* pe_bng = (const float*)d_in[4];
  const float* pe_bnb = (const float*)d_in[5];
  const float* pe_w2  = (const float*)d_in[6];
  const float* pe_b2  = (const float*)d_in[7];
  const float* ip_w1  = (const float*)d_in[8];
  const float* ip_b1  = (const float*)d_in[9];
  const float* ip_bng = (const float*)d_in[10];
  const float* ip_bnb = (const float*)d_in[11];
  const float* ip_w2  = (const float*)d_in[12];
  const float* ip_b2  = (const float*)d_in[13];
  const float* ln1_g  = (const float*)d_in[14];
  const float* ln1_b  = (const float*)d_in[15];
  const float* qkv_w  = (const float*)d_in[16];
  const float* proj_w = (const float*)d_in[17];
  const float* proj_b = (const float*)d_in[18];
  const float* ln2_g  = (const float*)d_in[19];
  const float* ln2_b  = (const float*)d_in[20];
  const float* fc1_w  = (const float*)d_in[21];
  const float* fc1_b  = (const float*)d_in[22];
  const float* fc2_w  = (const float*)d_in[23];
  const float* fc2_b  = (const float*)d_in[24];
  const float* knn_w  = (const float*)d_in[25];
  const float* knn_b  = (const float*)d_in[26];
  const float* mrg_w  = (const float*)d_in[27];
  const float* mrg_b  = (const float*)d_in[28];
  const float* id_w1  = (const float*)d_in[29];
  const float* id_b1  = (const float*)d_in[30];
  const float* id_bng = (const float*)d_in[31];
  const float* id_bnb = (const float*)d_in[32];
  const float* id_w2  = (const float*)d_in[33];
  const float* id_b2  = (const float*)d_in[34];
  const float* norm_g = (const float*)d_in[35];
  const float* norm_b = (const float*)d_in[36];
  float* out = (float*)d_out;
  (void)in_sizes; (void)n_in; (void)out_size; (void)ws_size;

  // ---- carve workspace ----
  uintptr_t p = (uintptr_t)d_ws;
  auto carveB = [&](size_t bytes) -> void* {
    void* r = (void*)p;
    p += (bytes + 255) & ~(size_t)255;
    return r;
  };
  auto cf = [&](size_t n) -> float* { return (float*)carveB(n * 4); };
  auto ch = [&](size_t n) -> u16*   { return (u16*)  carveB(n * 2); };

  // fp32 activations
  float* pos  = cf((size_t)M_ * C_);
  float* x    = cf((size_t)M_ * C_);
  float* xi   = cf((size_t)M_ * C_);
  float* x1   = cf((size_t)M_ * C_);
  float* t128 = cf((size_t)M_ * 128);
  float* sc   = cf((size_t)B_ * H_ * N_ * N_);
  float* g1   = cf((size_t)M_ * OUT_);
  float* g2   = cf((size_t)M_ * OUT_);
  float* gfb  = cf((size_t)B_ * OUT_);
  int*   kidx = (int*)carveB((size_t)B_ * K_ * N_ * 4);
  // aliases into regions dead at time of use
  float* t1 = g1;                 // layer-0 only; g1/g2 used post-loop
  float* t2 = g2;
  float* mo = sc;                 // scores fp32 dead after softmax

  // bf16 activations / mirrors
  u16* coorT   = ch((size_t)M_ * 32);
  u16* fT      = ch((size_t)M_ * 128);
  u16* t128b   = ch((size_t)M_ * 128);
  u16* nx_bf   = ch((size_t)M_ * C_);
  u16* qkv_bf  = ch((size_t)M_ * 3 * C_);
  u16* sc_bf   = ch((size_t)B_ * H_ * N_ * N_);
  u16* vT      = ch((size_t)B_ * H_ * HD_ * N_);
  u16* ao_bf   = ch((size_t)M_ * C_);
  u16* x1_bf   = ch((size_t)M_ * C_);
  u16* knnf_bf = ch((size_t)M_ * C_);
  u16* x_bf    = ch((size_t)M_ * C_);
  u16* g1_bf   = ch((size_t)M_ * OUT_);

  // bf16 weights
  u16* pe_w1p  = ch(128 * 32);
  u16* pe_w2b  = ch((size_t)C_ * 128);
  u16* ip_w1b  = ch((size_t)C_ * 128);
  u16* ip_w2b  = ch((size_t)C_ * C_);
  u16* qkv_wb  = ch((size_t)DEPTH_ * 3 * C_ * C_);
  u16* proj_wb = ch((size_t)DEPTH_ * C_ * C_);
  u16* fc1_wb  = ch((size_t)DEPTH_ * HID_ * C_);
  u16* fc2_wb  = ch((size_t)DEPTH_ * C_ * HID_);
  u16* knn_wb  = ch((size_t)C_ * 2 * C_);
  u16* wdb     = ch((size_t)C_ * C_);
  u16* mrg_wb  = ch((size_t)C_ * 2 * C_);
  u16* id_w1b  = ch((size_t)OUT_ * C_);
  u16* id_w2b  = ch((size_t)OUT_ * OUT_);

  const size_t nMC = (size_t)M_ * C_;

  // ---- weight casts (bf16 mirrors) ----
  pct_pad_w1<<<blk1(128 * 32), 256, 0, stream>>>(pe_w1, pe_w1p);
  castw(stream, pe_w2, pe_w2b, (size_t)C_ * 128);
  castw(stream, ip_w1, ip_w1b, (size_t)C_ * 128);
  castw(stream, ip_w2, ip_w2b, (size_t)C_ * C_);
  castw(stream, qkv_w, qkv_wb, (size_t)DEPTH_ * 3 * C_ * C_);
  castw(stream, proj_w, proj_wb, (size_t)DEPTH_ * C_ * C_);
  castw(stream, fc1_w, fc1_wb, (size_t)DEPTH_ * HID_ * C_);
  castw(stream, fc2_w, fc2_wb, (size_t)DEPTH_ * C_ * HID_);
  castw(stream, knn_w, knn_wb, (size_t)C_ * 2 * C_);
  castw(stream, mrg_w, mrg_wb, (size_t)C_ * 2 * C_);
  castw(stream, id_w1, id_w1b, (size_t)OUT_ * C_);
  castw(stream, id_w2, id_w2b, (size_t)OUT_ * OUT_);
  pct_wd_prep<<<blk1((size_t)C_ * C_), 256, 0, stream>>>(knn_w, wdb);

  // ---- preprocessing ----
  pct_transpose_pad<<<blk1((size_t)M_ * 32), 256, 0, stream>>>(coor, coorT, 3, 32);
  pct_transpose_pad<<<blk1((size_t)M_ * 128), 256, 0, stream>>>(f_in, fT, 128, 128);
  pct_knn_topk<<<dim3(M_ / 256), 256, 0, stream>>>(coor, kidx);

  // ---- positional embed: conv(3->128) -> bn+lrelu -> conv(128->768) ----
  run_gemm(stream, coorT, 32, pe_w1p, 32, t128, nullptr, 128, pe_b1, M_, 128, 32);
  pct_bn_lrelu<<<blk1((size_t)M_ * 128), 256, 0, stream>>>(t128, t128b, pe_bng, pe_bnb, 128);
  run_gemm(stream, t128b, 128, pe_w2b, 128, pos, nullptr, C_, pe_b2, M_, C_, 128);

  // ---- input embed: conv(128->768) -> bn+lrelu -> conv(768->768) ----
  run_gemm(stream, fT, 128, ip_w1b, 128, x1, nullptr, C_, ip_b1, M_, C_, 128);
  pct_bn_lrelu<<<blk1(nMC), 256, 0, stream>>>(x1, nx_bf, ip_bng, ip_bnb, C_);
  run_gemm(stream, nx_bf, C_, ip_w2b, C_, x, nullptr, C_, ip_b2, M_, C_, C_);

  // ---- transformer layers ----
  for (int i = 0; i < DEPTH_; ++i) {
    pct_add<<<blk1(nMC), 256, 0, stream>>>(xi, x, pos, nMC, nullptr);
    pct_ln_row<<<dim3(M_), 256, 0, stream>>>(xi, nullptr, nx_bf,
                                             ln1_g + (size_t)i * C_,
                                             ln1_b + (size_t)i * C_, C_);
    // qkv (bf16-only output)
    run_gemm(stream, nx_bf, C_, qkv_wb + (size_t)i * 3 * C_ * C_, C_,
             nullptr, qkv_bf, 3 * C_, nullptr, M_, 3 * C_, C_);
    pct_vt<<<blk1((size_t)B_ * H_ * HD_ * N_), 256, 0, stream>>>(qkv_bf, vT);
    // scores: per (b,h)  Q(512x128) x K(512x128)^T, scaled -> fp32
    run_gemm(stream, qkv_bf, 3 * C_, qkv_bf + C_, 3 * C_, sc, nullptr, N_, nullptr,
             N_, N_, HD_, 0, 0, 0.08838834764831845f, B_ * H_, H_,
             (long long)N_ * 3 * C_, HD_, (long long)N_ * 3 * C_, HD_,
             (long long)H_ * N_ * N_, (long long)N_ * N_);
    pct_softmax_row<<<dim3(B_ * H_ * N_), 256, 0, stream>>>(sc, sc_bf, N_);
    // O: per (b,h)  A(512x512) x vT^T  -> bf16-only output (heads concat)
    run_gemm(stream, sc_bf, N_, vT, N_, nullptr, ao_bf, C_, nullptr,
             N_, HD_, N_, 0, 0, 1.0f, B_ * H_, H_,
             (long long)H_ * N_ * N_, (long long)N_ * N_,
             (long long)H_ * HD_ * N_, (long long)HD_ * N_,
             (long long)N_ * C_, HD_);
    // projection -> fp32 + bf16 mirror
    run_gemm(stream, ao_bf, C_, proj_wb + (size_t)i * C_ * C_, C_,
             x1, x1_bf, C_, proj_b + (size_t)i * C_, M_, C_, C_);

    float* res = x1;
    if (i < 1) {   // KNN_LAYER == 1
      run_gemm(stream, nx_bf, C_, knn_wb, 2 * C_, t1, nullptr, C_, nullptr, M_, C_, C_);
      run_gemm(stream, nx_bf, C_, wdb, C_, t2, nullptr, C_, knn_b, M_, C_, C_);
      pct_knn_gather_max<<<blk1(nMC), 256, 0, stream>>>(t1, t2, kidx, knnf_bf);
      run_gemm(stream, x1_bf, C_, mrg_wb, 2 * C_, mo, nullptr, C_, mrg_b, M_, C_, C_);
      run_gemm(stream, knnf_bf, C_, mrg_wb + C_, 2 * C_, mo, nullptr, C_, nullptr,
               M_, C_, C_, 0, /*accum=*/1);
      res = mo;
    }
    pct_add<<<blk1(nMC), 256, 0, stream>>>(xi, xi, res, nMC, nullptr);

    pct_ln_row<<<dim3(M_), 256, 0, stream>>>(xi, nullptr, nx_bf,
                                             ln2_g + (size_t)i * C_,
                                             ln2_b + (size_t)i * C_, C_);
    // MLP: fc1 (gelu, bf16-only out into qkv_bf) then fc2
    run_gemm(stream, nx_bf, C_, fc1_wb + (size_t)i * HID_ * C_, C_,
             nullptr, qkv_bf, HID_, fc1_b + (size_t)i * HID_, M_, HID_, C_, /*act=*/2);
    run_gemm(stream, qkv_bf, HID_, fc2_wb + (size_t)i * C_ * HID_, HID_,
             x1, nullptr, C_, fc2_b + (size_t)i * C_, M_, C_, HID_);
    pct_add<<<blk1(nMC), 256, 0, stream>>>(x, xi, x1, nMC,
                                           (i == DEPTH_ - 1) ? x_bf : nullptr);
  }

  // ---- head: conv(768->1024) -> bn+lrelu -> conv(1024->1024) -> maxN -> LN ----
  run_gemm(stream, x_bf, C_, id_w1b, C_, g1, nullptr, OUT_, id_b1, M_, OUT_, C_);
  pct_bn_lrelu<<<blk1((size_t)M_ * OUT_), 256, 0, stream>>>(g1, g1_bf, id_bng, id_bnb, OUT_);
  run_gemm(stream, g1_bf, OUT_, id_w2b, OUT_, g2, nullptr, OUT_, id_b2, M_, OUT_, OUT_);
  pct_max_over_n<<<blk1((size_t)B_ * OUT_), 256, 0, stream>>>(g2, gfb);
  pct_ln_row<<<dim3(B_), 256, 0, stream>>>(gfb, out, nullptr, norm_g, norm_b, OUT_);
}
